// PerformerBlock_89876485636715
// MI455X (gfx1250) — matrix operand order repaired and checked by reading, not compiled, not run on hardware
//
#include <hip/hip_runtime.h>
#include <hip/hip_bf16.h>
#include <math.h>

typedef __bf16 bf16;
typedef __attribute__((ext_vector_type(16))) __bf16 v16bf;
typedef __attribute__((ext_vector_type(8)))  __bf16 v8bf;
typedef __attribute__((ext_vector_type(8)))  float  v8f;
typedef __attribute__((ext_vector_type(4)))  int    v4i;

#define AS1 __attribute__((address_space(1)))
#define AS3 __attribute__((address_space(3)))

#if defined(__has_builtin)
# if __has_builtin(__builtin_amdgcn_global_load_async_to_lds_b128) && \
     __has_builtin(__builtin_amdgcn_s_wait_asynccnt)
#  define USE_ASYNC_COPY 1
# endif
#endif
#ifndef USE_ASYNC_COPY
# define USE_ASYNC_COPY 0
#endif

#define BATCH 4
#define SEQ   4096
#define EDIM  512
#define NHEAD 8
#define HDIM  64
#define NFEAT 256
#define MLPH  1024
#define MTOT  (BATCH*SEQ)

#define BM 64
#define BK 32

__device__ __forceinline__ v16bf cat8(v8bf lo, v8bf hi) {
  return __builtin_shufflevector(lo, hi, 0,1,2,3,4,5,6,7,8,9,10,11,12,13,14,15);
}

#if USE_ASYNC_COPY
__device__ __forceinline__ void async_copy16(const bf16* g, bf16* l) {
  __builtin_amdgcn_global_load_async_to_lds_b128((AS1 v4i*)g, (AS3 v4i*)l, 0, 0);
}
#endif

// ---------------------------------------------------------------------------
// Generic wave32 bf16 WMMA GEMM:  C = epi(A @ B), block tile BM x BNT x BK.
//   A: [M,K] bf16 row-major (lda), or transposed view (A[m,k] = mem[k*lda+m])
//   B: [K,N] bf16 row-major (ldb)
//   batched over gridDim.z with quotient/remainder strides.
// 8 waves arranged 4(m) x 2(n-groups); each wave owns a 16 x (BNT/2) slab:
//   NT = BNT/32 accumulators, one shared A fragment -> NT WMMAs per k-step.
// LDS layouts:
//   sA: row-major [BM][BK]  -> A frag = 2x ds_load_b128 per lane
//   sB: fragment-major swizzle [ntile][lane][e] -> B frag = 2x ds_load_b128
// A tile staged with gfx1250 global_load_async_to_lds_b128 when available.
// ---------------------------------------------------------------------------
template<int BNT, bool TRANS_A, bool HAS_BIAS, bool HAS_RES, bool GELU_ACT,
         bool ROW_SCALE, bool STORE_F32, bool STORE_BF16>
__global__ __launch_bounds__(256) void gemm_bf16_wmma(
    const bf16* __restrict__ A, int lda, int divA, long long sqA, long long srA,
    const bf16* __restrict__ Bm, int ldb, int divB, long long sqB, long long srB,
    const float* __restrict__ bias,
    const float* __restrict__ res, int ldres, long long sres,
    const float* __restrict__ rscale, long long sscale,
    float* __restrict__ Cf, bf16* __restrict__ Cb,
    int ldc, int divC, long long sqC, long long srC,
    int Mdim, int Ndim, int Kdim)
{
  (void)Mdim; (void)Ndim;
  constexpr int NT = BNT / 32;           // WMMA n-tiles (and accs) per wave
  __shared__ bf16 sA[BM * BK];
  __shared__ bf16 sB[BK * BNT];          // swizzled: [(nt*32+lane)*16 + e]

  const int z = blockIdx.z;
  A  += (long long)(z / divA) * sqA + (long long)(z % divA) * srA;
  Bm += (long long)(z / divB) * sqB + (long long)(z % divB) * srB;
  const long long coff = (long long)(z / divC) * sqC + (long long)(z % divC) * srC;

  const int m0 = blockIdx.y * BM;
  const int n0 = blockIdx.x * BNT;
  const int tid  = threadIdx.x;
  const int lane = tid & 31;
  const int wave = tid >> 5;
  const int mo   = (wave & 3) * 16;      // wave's row-tile inside block
  const int ngrp = wave >> 2;            // wave's n-group (covers BNT/2 cols)
  const int r    = lane & 15;
  const int half = lane >> 4;

  v8f acc[NT] = {};

  const int nk = Kdim / BK;
  for (int kb = 0; kb < nk; ++kb) {
    const int k0 = kb * BK;

    // ---- stage A tile (64x32 row-major) ----
    if (!TRANS_A) {
#if USE_ASYNC_COPY
      {
        int row = tid >> 2, q = tid & 3;        // 4x 16B chunks per 64B row
        async_copy16(A + (long long)(m0 + row) * lda + k0 + q * 8,
                     sA + row * BK + q * 8);
      }
#else
      #pragma unroll
      for (int i = 0; i < 4; ++i) {
        int idx = tid * 4 + i;
        int row = idx >> 4;
        int c2  = idx & 15;
        ((uint32_t*)sA)[idx] =
            *(const uint32_t*)(A + (long long)(m0 + row) * lda + k0 + c2 * 2);
      }
#endif
    } else {
      // element-wise strided gather of A^T into row-major sA
      #pragma unroll
      for (int i = 0; i < 8; ++i) {
        int idx = tid * 8 + i;
        int row = idx >> 5;
        int col = idx & 31;
        sA[idx] = A[(long long)(k0 + col) * lda + (m0 + row)];
      }
    }

    // ---- stage B tile swizzled: 8 strided u16 loads -> 1 ds_store_b128 ----
    #pragma unroll
    for (int j = 0; j < BNT / 64; ++j) {
      int q     = j * 256 + tid;
      int btile = q >> 6;
      int bl    = (q >> 1) & 31;
      int beh   = q & 1;
      int bcol  = n0 + btile * 16 + (bl & 15);
      int bkoff = (bl >> 4) * 8 + beh * 16;
      v8bf tmp;
      #pragma unroll
      for (int e = 0; e < 8; ++e)
        tmp[e] = Bm[(long long)(k0 + bkoff + e) * ldb + bcol];
      *(v8bf*)(sB + (btile * 32 + bl) * 16 + beh * 8) = tmp;
    }

#if USE_ASYNC_COPY
    if (!TRANS_A) __builtin_amdgcn_s_wait_asynccnt(0);
#endif
    __syncthreads();

    if (kb + 1 < nk) {  // gfx1250 global_prefetch_b8 of next A tile
      const bf16* nexta = TRANS_A ? (A + (long long)(k0 + BK) * lda + m0)
                                  : (A + (long long)(m0 + (tid & 63)) * lda + k0 + BK);
      __builtin_prefetch(nexta, 0, 1);
    }

    // ---- fragments: all contiguous 16B LDS loads ----
    const bf16* ab = sA + (mo + r) * BK + half * 8;
    v16bf af = cat8(*(const v8bf*)ab, *(const v8bf*)(ab + 16));

    v16bf bfr[NT];
    #pragma unroll
    for (int i = 0; i < NT; ++i) {
      const bf16* bp = sB + ((ngrp * NT + i) * 32 + lane) * 16;
      bfr[i] = cat8(*(const v8bf*)bp, *(const v8bf*)(bp + 8));
    }
    #pragma unroll
    for (int i = 0; i < NT; ++i)
      acc[i] = __builtin_amdgcn_wmma_f32_16x16x32_bf16(false, af, false, bfr[i],
                                                       (short)0, acc[i], false, false);
    __syncthreads();
  }

  // ---- epilogue: C/D layout lane<16:(M=e,N=lane) lane>=16:(M=8+e,N=lane-16) ----
  #pragma unroll
  for (int e = 0; e < 8; ++e) {
    int row = m0 + mo + half * 8 + e;
    #pragma unroll
    for (int i = 0; i < NT; ++i) {
      int col = n0 + ngrp * (BNT / 2) + i * 16 + r;
      float v = acc[i][e];
      if (HAS_BIAS)  v += bias[col];
      if (GELU_ACT)  v = 0.5f * v * (1.0f + erff(v * 0.70710678118f));
      if (HAS_RES)   v += res[(long long)z * sres + (long long)row * ldres + col];
      if (ROW_SCALE) v *= rscale[(long long)z * sscale + row];
      long long o = coff + (long long)row * ldc + col;
      if (STORE_F32)  Cf[o] = v;
      if (STORE_BF16) Cb[o] = (bf16)v;
    }
  }
}

// ---------------------------------------------------------------------------
// Performer random-feature map: per (b,h) row-block of 16 rows,
//   xp = q_head @ Pmat  (WMMA, K=64, full 256 cols in one block)
//   out = exp(xp - rowmax) / 16
// sB holds Pmat in the same fragment-major swizzle: [((ks*16+nt)*32+lane)*16+e]
// ---------------------------------------------------------------------------
__global__ __launch_bounds__(256) void performer_feature(
    const bf16* __restrict__ qkv, int qoff,
    const bf16* __restrict__ Pm,          // [HDIM, NFEAT] bf16
    bf16* __restrict__ outp)              // [B*H, SEQ, NFEAT] bf16
{
  __shared__ __align__(16) unsigned char smem[16 * HDIM * 2 + HDIM * NFEAT * 2];
  bf16*  sA   = (bf16*)smem;                       // 16 x 64 row-major
  bf16*  sB   = (bf16*)(smem + 16 * HDIM * 2);     // swizzled Pmat
  float* sOut = (float*)(smem + 16 * HDIM * 2);    // 16 x 256, aliases sB
  __shared__ float smax[16];

  const int bh = blockIdx.z;
  const int b  = bh >> 3;
  const int h  = bh & 7;
  const int brow = b * SEQ + blockIdx.y * 16;

  const int tid  = threadIdx.x;
  const int lane = tid & 31;
  const int wave = tid >> 5;
  const int r    = lane & 15;
  const int half = lane >> 4;

  // stage A: 16x64 = 128 x 16B chunks
  const bf16* qbase = qkv + (long long)brow * (3 * EDIM) + qoff + h * HDIM;
#if USE_ASYNC_COPY
  if (tid < 128) {
    int row = tid >> 3, q = tid & 7;
    async_copy16(qbase + (long long)row * (3 * EDIM) + q * 8,
                 sA + row * HDIM + q * 8);
  }
#else
  #pragma unroll
  for (int i = 0; i < 2; ++i) {
    int idx = tid * 2 + i;
    int row = idx >> 5;
    int c2  = idx & 31;
    ((uint32_t*)sA)[idx] =
        *(const uint32_t*)(qbase + (long long)row * (3 * EDIM) + c2 * 2);
  }
#endif

  // stage Pmat swizzled: 2048 chunks of 8 elements, 8 per thread
  #pragma unroll
  for (int j = 0; j < 8; ++j) {
    int q    = j * 256 + tid;
    int eh   = q & 1;
    int l    = (q >> 1) & 31;
    int nt   = (q >> 6) & 15;
    int ks   = q >> 10;
    int col  = nt * 16 + (l & 15);
    int krow = ks * 32 + (l >> 4) * 8 + eh * 16;
    v8bf tmp;
    #pragma unroll
    for (int e = 0; e < 8; ++e)
      tmp[e] = Pm[(krow + e) * NFEAT + col];
    *(v8bf*)(sB + ((ks * 16 + nt) * 32 + l) * 16 + eh * 8) = tmp;
  }

#if USE_ASYNC_COPY
  __builtin_amdgcn_s_wait_asynccnt(0);
#endif
  __syncthreads();

  v8f acc0 = {};
  v8f acc1 = {};
  const int nt0 = wave * 2, nt1 = wave * 2 + 1;
  #pragma unroll
  for (int ks = 0; ks < 2; ++ks) {
    const bf16* ab = sA + r * HDIM + ks * 32 + half * 8;
    v16bf af = cat8(*(const v8bf*)ab, *(const v8bf*)(ab + 16));
    const bf16* b0 = sB + ((ks * 16 + nt0) * 32 + lane) * 16;
    const bf16* b1 = sB + ((ks * 16 + nt1) * 32 + lane) * 16;
    v16bf bfr0 = cat8(*(const v8bf*)b0, *(const v8bf*)(b0 + 8));
    v16bf bfr1 = cat8(*(const v8bf*)b1, *(const v8bf*)(b1 + 8));
    acc0 = __builtin_amdgcn_wmma_f32_16x16x32_bf16(false, af, false, bfr0,
                                                   (short)0, acc0, false, false);
    acc1 = __builtin_amdgcn_wmma_f32_16x16x32_bf16(false, af, false, bfr1,
                                                   (short)0, acc1, false, false);
  }
  __syncthreads();   // everyone done reading sB before we alias it as sOut

  #pragma unroll
  for (int e = 0; e < 8; ++e) {
    sOut[(half * 8 + e) * NFEAT + nt0 * 16 + r] = acc0[e];
    sOut[(half * 8 + e) * NFEAT + nt1 * 16 + r] = acc1[e];
  }
  __syncthreads();

  if (tid < 16) {
    float mx = -3.4e38f;
    for (int c = 0; c < NFEAT; ++c) mx = fmaxf(mx, sOut[tid * NFEAT + c]);
    smax[tid] = mx;
  }
  __syncthreads();

  for (int i = 0; i < 16; ++i) {
    float v = expf(sOut[i * NFEAT + tid] - smax[i]) * 0.0625f;  // 1/sqrt(256)
    outp[((long long)bh * SEQ + blockIdx.y * 16 + i) * NFEAT + tid] = (bf16)v;
  }
}

// ---------------------------------------------------------------------------
// LayerNorm over EDIM=512, one wave per row, output bf16
// ---------------------------------------------------------------------------
__global__ __launch_bounds__(256) void layernorm_bf16(
    const float* __restrict__ x, const float* __restrict__ g,
    const float* __restrict__ be, bf16* __restrict__ out)
{
  int row  = blockIdx.x * 8 + (threadIdx.x >> 5);
  int lane = threadIdx.x & 31;
  const float* xr = x + (long long)row * EDIM;
  float v[16];
  float s = 0.f;
  #pragma unroll
  for (int i = 0; i < 16; ++i) { v[i] = xr[lane + i * 32]; s += v[i]; }
  #pragma unroll
  for (int o = 16; o; o >>= 1) s += __shfl_xor(s, o, 32);
  float mu = s * (1.0f / EDIM);
  float s2 = 0.f;
  #pragma unroll
  for (int i = 0; i < 16; ++i) { float d = v[i] - mu; s2 += d * d; }
  #pragma unroll
  for (int o = 16; o; o >>= 1) s2 += __shfl_xor(s2, o, 32);
  float rstd = rsqrtf(s2 * (1.0f / EDIM) + 1e-5f);
  #pragma unroll
  for (int i = 0; i < 16; ++i) {
    int c = lane + i * 32;
    out[(long long)row * EDIM + c] = (bf16)((v[i] - mu) * rstd * g[c] + be[c]);
  }
}

__global__ void f32_to_bf16(const float* __restrict__ in, bf16* __restrict__ out, int n) {
  int i = blockIdx.x * 256 + threadIdx.x;
  if (i < n) out[i] = (bf16)in[i];
}

// k_sum[bh][f] = sum_n k'[bh][n][f] + 1e-6
__global__ __launch_bounds__(256) void ksum_kernel(
    const bf16* __restrict__ kp, float* __restrict__ ksum)
{
  int bh = blockIdx.x, f = threadIdx.x;
  const bf16* p = kp + (long long)bh * SEQ * NFEAT + f;
  float s = 1e-6f;
  for (int n = 0; n < SEQ; ++n) s += (float)p[(long long)n * NFEAT];
  ksum[bh * NFEAT + f] = s;
}

// z[row] = 1 / dot(q'[row], k_sum[bh])
__global__ __launch_bounds__(256) void zinv_kernel(
    const bf16* __restrict__ qp, const float* __restrict__ ksum,
    float* __restrict__ z)
{
  long long row = (long long)blockIdx.x * 256 + threadIdx.x;
  int bh = (int)(row >> 12);
  const bf16* p = qp + row * NFEAT;
  const float* ks = ksum + bh * NFEAT;
  float s = 0.f;
  for (int f = 0; f < NFEAT; ++f) s += (float)p[f] * ks[f];
  z[row] = 1.0f / s;
}

// ---------------------------------------------------------------------------
extern "C" void kernel_launch(void* const* d_in, const int* in_sizes, int n_in,
                              void* d_out, int out_size, void* d_ws, size_t ws_size,
                              hipStream_t stream) {
  (void)in_sizes; (void)n_in; (void)out_size; (void)ws_size;
  const float* x     = (const float*)d_in[0];
  const float* Wqkv  = (const float*)d_in[1];
  const float* bqkv  = (const float*)d_in[2];
  const float* Pmat  = (const float*)d_in[3];
  const float* Wproj = (const float*)d_in[4];
  const float* bproj = (const float*)d_in[5];
  const float* W1    = (const float*)d_in[6];
  const float* b1    = (const float*)d_in[7];
  const float* W2    = (const float*)d_in[8];
  const float* b2    = (const float*)d_in[9];
  const float* g1    = (const float*)d_in[10];
  const float* be1   = (const float*)d_in[11];
  const float* g2    = (const float*)d_in[12];
  const float* be2   = (const float*)d_in[13];
  float* out = (float*)d_out;

  char* wsb = (char*)d_ws;
  size_t off = 0;
  auto carve = [&](size_t bytes) -> void* {
    off = (off + 255) & ~size_t(255);
    void* p = wsb + off;
    off += bytes;
    return p;
  };
  bf16*  h_bf    = (bf16*)carve((size_t)MTOT * EDIM * 2);
  bf16*  wqkv_bf = (bf16*)carve((size_t)EDIM * 3 * EDIM * 2);
  bf16*  wproj_bf= (bf16*)carve((size_t)EDIM * EDIM * 2);
  bf16*  w1_bf   = (bf16*)carve((size_t)EDIM * MLPH * 2);
  bf16*  w2_bf   = (bf16*)carve((size_t)MLPH * EDIM * 2);
  bf16*  pmat_bf = (bf16*)carve((size_t)HDIM * NFEAT * 2);
  bf16*  qkv_bf  = (bf16*)carve((size_t)MTOT * 3 * EDIM * 2);
  bf16*  qp      = (bf16*)carve((size_t)BATCH * NHEAD * SEQ * NFEAT * 2);
  bf16*  kp      = (bf16*)carve((size_t)BATCH * NHEAD * SEQ * NFEAT * 2);
  bf16*  kv_bf   = (bf16*)carve((size_t)BATCH * NHEAD * NFEAT * HDIM * 2);
  float* ksum    = (float*)carve((size_t)BATCH * NHEAD * NFEAT * 4);
  float* zbuf    = (float*)carve((size_t)BATCH * NHEAD * SEQ * 4);
  bf16*  attn_bf = (bf16*)carve((size_t)MTOT * EDIM * 2);
  float* x1      = (float*)carve((size_t)MTOT * EDIM * 4);
  bf16*  h2_bf   = (bf16*)carve((size_t)MTOT * EDIM * 2);
  bf16*  mlp1_bf = (bf16*)carve((size_t)MTOT * MLPH * 2);

  // weights -> bf16
  f32_to_bf16<<<(EDIM*3*EDIM + 255)/256, 256, 0, stream>>>(Wqkv, wqkv_bf, EDIM*3*EDIM);
  f32_to_bf16<<<(HDIM*NFEAT  + 255)/256, 256, 0, stream>>>(Pmat, pmat_bf, HDIM*NFEAT);
  f32_to_bf16<<<(EDIM*EDIM   + 255)/256, 256, 0, stream>>>(Wproj, wproj_bf, EDIM*EDIM);
  f32_to_bf16<<<(EDIM*MLPH   + 255)/256, 256, 0, stream>>>(W1, w1_bf, EDIM*MLPH);
  f32_to_bf16<<<(MLPH*EDIM   + 255)/256, 256, 0, stream>>>(W2, w2_bf, MLPH*EDIM);

  // LN1
  layernorm_bf16<<<MTOT/8, 256, 0, stream>>>(x, g1, be1, h_bf);

  // qkv = h @ Wqkv + bqkv  -> bf16 [MTOT, 1536]   (tile 64x128)
  gemm_bf16_wmma<128,false,true,false,false,false,false,true>
      <<<dim3(3*EDIM/128, MTOT/BM, 1), 256, 0, stream>>>(
      h_bf, EDIM, 1, 0, 0,
      wqkv_bf, 3*EDIM, 1, 0, 0,
      bqkv, nullptr, 0, 0, nullptr, 0,
      nullptr, qkv_bf, 3*EDIM, 1, 0, 0,
      MTOT, 3*EDIM, EDIM);

  // feature maps q' and k'
  performer_feature<<<dim3(1, SEQ/16, BATCH*NHEAD), 256, 0, stream>>>(
      qkv_bf, 0, pmat_bf, qp);
  performer_feature<<<dim3(1, SEQ/16, BATCH*NHEAD), 256, 0, stream>>>(
      qkv_bf, EDIM, pmat_bf, kp);

  // kv[bh] = k'^T @ v   (M=256, N=64, K=4096), v is strided view of qkv
  gemm_bf16_wmma<64,true,false,false,false,false,false,true>
      <<<dim3(HDIM/64, NFEAT/BM, BATCH*NHEAD), 256, 0, stream>>>(
      kp, NFEAT, 1, (long long)SEQ*NFEAT, 0,
      qkv_bf + 2*EDIM, 3*EDIM, NHEAD, (long long)SEQ*3*EDIM, HDIM,
      nullptr, nullptr, 0, 0, nullptr, 0,
      nullptr, kv_bf, HDIM, 1, (long long)NFEAT*HDIM, 0,
      NFEAT, HDIM, SEQ);

  ksum_kernel<<<BATCH*NHEAD, NFEAT, 0, stream>>>(kp, ksum);
  zinv_kernel<<<BATCH*NHEAD*SEQ/256, 256, 0, stream>>>(qp, ksum, zbuf);

  // attn[bh] = (q' @ kv) * z   -> bf16 laid out [b, n, h*64+e]
  gemm_bf16_wmma<64,false,false,false,false,true,false,true>
      <<<dim3(HDIM/64, SEQ/BM, BATCH*NHEAD), 256, 0, stream>>>(
      qp, NFEAT, 1, (long long)SEQ*NFEAT, 0,
      kv_bf, HDIM, 1, (long long)NFEAT*HDIM, 0,
      nullptr, nullptr, 0, 0, zbuf, SEQ,
      nullptr, attn_bf, EDIM, NHEAD, (long long)SEQ*EDIM, HDIM,
      SEQ, HDIM, NFEAT);

  // x1 = x + attn @ Wproj + bproj  -> f32   (tile 64x128)
  gemm_bf16_wmma<128,false,true,true,false,false,true,false>
      <<<dim3(EDIM/128, MTOT/BM, 1), 256, 0, stream>>>(
      attn_bf, EDIM, 1, 0, 0,
      wproj_bf, EDIM, 1, 0, 0,
      bproj, x, EDIM, 0, nullptr, 0,
      x1, nullptr, EDIM, 1, 0, 0,
      MTOT, EDIM, EDIM);

  // LN2
  layernorm_bf16<<<MTOT/8, 256, 0, stream>>>(x1, g2, be2, h2_bf);

  // mlp1 = gelu(h2 @ W1 + b1) -> bf16   (tile 64x128)
  gemm_bf16_wmma<128,false,true,false,true,false,false,true>
      <<<dim3(MLPH/128, MTOT/BM, 1), 256, 0, stream>>>(
      h2_bf, EDIM, 1, 0, 0,
      w1_bf, MLPH, 1, 0, 0,
      b1, nullptr, 0, 0, nullptr, 0,
      nullptr, mlp1_bf, MLPH, 1, 0, 0,
      MTOT, MLPH, EDIM);

  // out = x1 + mlp1 @ W2 + b2 -> f32 (final output)   (tile 64x128)
  gemm_bf16_wmma<128,false,true,true,false,false,true,false>
      <<<dim3(EDIM/128, MTOT/BM, 1), 256, 0, stream>>>(
      mlp1_bf, MLPH, 1, 0, 0,
      w2_bf, EDIM, 1, 0, 0,
      b2, x1, EDIM, 0, nullptr, 0,
      out, nullptr, EDIM, 1, 0, 0,
      MTOT, EDIM, MLPH);
}
